// Decoder_19550691131855
// MI455X (gfx1250) — matrix operand order repaired and checked by reading, not compiled, not run on hardware
//
#include <hip/hip_runtime.h>
#include <stdint.h>

// ---------------- problem constants ----------------
#define DIM   128
#define HID   512
#define NOUT  100000
#define NSEG  5
#define SEG   20000        // NOUT / NSEG
#define WIN   10000        // NOUT / 10
#define MUF   5000.0f      // WIN / 2

// ---------------- WMMA conv tiling ----------------
#define KITER       313    // k0 = 32*j, j in [0,313): covers taps [0,10000) for all m rows
#define SPAD_STRIDE 30720  // per-segment padded signal stride (bf16 elements), >= 30480 used
#define PADF        5000   // front zero padding of each segment

// ---------------- workspace layout (bytes) ----------------
#define WS_H      0                      // 512 f32
#define WS_SIG    2048                   // 5 f32
#define WS_WSUM   2176                   // 5 f32
#define WS_SPAD   2304                   // 5*30720 u16 (bf16)      = 307200 B
#define WS_STAGEA 309504                 // 5*313*32*16 u16 (bf16)  = 1602560 B
// total ~1.91 MB of d_ws

typedef __attribute__((ext_vector_type(16))) __bf16 v16bf;
typedef __attribute__((ext_vector_type(8)))  float  v8f;

union BF16x16 { v16bf v; uint4  u[2]; unsigned short s[16]; };
union F32x8   { v8f   v; float4 f4[2]; };

// round-to-nearest-even f32 -> bf16 (bit twiddle; no __bf16 arithmetic needed)
static __device__ __forceinline__ unsigned short f2bf(float f) {
  unsigned int u = __float_as_uint(f);
  u += 0x7FFFu + ((u >> 16) & 1u);
  return (unsigned short)(u >> 16);
}

// ---------------------------------------------------------------
// 0) zero the padded-signal region (pads must be 0; d_ws is poisoned)
__global__ void k_zero(unsigned int* __restrict__ p, int n32) {
  int i = blockIdx.x * blockDim.x + threadIdx.x;
  if (i < n32) p[i] = 0u;
}

// ---------------------------------------------------------------
// 1) head: sigmas = x@W_sig + b_sig ; h = swish(x@W1 + b1)
__global__ void k_head(const float* __restrict__ x,
                       const float* __restrict__ W_sig,
                       const float* __restrict__ b_sig,
                       const float* __restrict__ W1,
                       const float* __restrict__ b1,
                       float* __restrict__ h,
                       float* __restrict__ sig) {
  __shared__ float xs[DIM];
  if (threadIdx.x < DIM) xs[threadIdx.x] = x[threadIdx.x];
  __syncthreads();
  if (threadIdx.x < NSEG) {
    float a = b_sig[threadIdx.x];
    for (int d = 0; d < DIM; ++d) a += xs[d] * W_sig[d * NSEG + threadIdx.x];
    sig[threadIdx.x] = a;
  }
  for (int j = threadIdx.x; j < HID; j += blockDim.x) {
    float a = b1[j];
    for (int d = 0; d < DIM; ++d) a += xs[d] * W1[d * HID + j];
    h[j] = a / (1.0f + __expf(-a));      // swish
  }
}

// ---------------------------------------------------------------
// 2) per-segment gaussian window normalization sum
__global__ void k_wsum(const float* __restrict__ sig, float* __restrict__ wsum) {
  __shared__ float red[256];
  const int sid = blockIdx.x;
  const float s   = sig[sid];
  const float inv = 1.0f / (2.0f * s * s);
  float acc = 0.0f;
  for (int t = threadIdx.x; t < WIN; t += 256) {
    float d = (float)t - MUF;
    acc += __expf(-d * d * inv);
  }
  red[threadIdx.x] = acc;
  __syncthreads();
  for (int off = 128; off > 0; off >>= 1) {
    if (threadIdx.x < off) red[threadIdx.x] += red[threadIdx.x + off];
    __syncthreads();
  }
  if (threadIdx.x == 0) wsum[sid] = red[0];
}

// ---------------------------------------------------------------
// 3) pre-stage WMMA A-fragments (window side), lane-major so the conv
//    loop reads 32 aligned bytes per lane per wmma.
//    A[m][K] = wrev[k0 - m + K], wrev[t] = wnorm[WIN-1-t], zero outside [0,WIN).
//    16-bit A layout: lane = m + 16*hi ; element e -> K = e + 16*hi
__global__ void k_stageA(const float* __restrict__ sig,
                         const float* __restrict__ wsum,
                         unsigned short* __restrict__ stagedA) {
  const int TOT = NSEG * KITER * 32;
  const int gid = blockIdx.x * blockDim.x + threadIdx.x;
  if (gid >= TOT) return;
  const int sid  = gid / (KITER * 32);
  const int rem  = gid - sid * (KITER * 32);
  const int j    = rem >> 5;
  const int lane = rem & 31;
  const int m    = lane & 15;
  const int hi   = lane >> 4;
  const int k0   = j << 5;
  const float s      = sig[sid];
  const float inv    = 1.0f / (2.0f * s * s);
  const float invsum = 1.0f / wsum[sid];
  BF16x16 frag;
  #pragma unroll
  for (int e = 0; e < 16; ++e) {
    const int K   = e + 16 * hi;
    const int tap = k0 - m + K;
    float v = 0.0f;
    if (tap >= 0 && tap < WIN) {
      float t = (float)(WIN - 1 - tap) - MUF;   // reversed window index
      v = __expf(-t * t * inv) * invsum;
    }
    frag.s[e] = f2bf(v);
  }
  uint4* dst = (uint4*)(stagedA + ((size_t)gid << 4));
  dst[0] = frag.u[0];
  dst[1] = frag.u[1];
}

// ---------------------------------------------------------------
// 4) GEMV y = h@W2 + b2 -> bf16 into padded signal buffer.
//    Bandwidth-bound: 204.8 MB of W2 streamed once, float4 coalesced
//    (~9 us floor at 23.3 TB/s).
__global__ void k_gemv(const float* __restrict__ h,
                       const float* __restrict__ W2,
                       const float* __restrict__ b2,
                       unsigned short* __restrict__ spad) {
  __shared__ float hl[HID];
  for (int i = threadIdx.x; i < HID; i += blockDim.x) hl[i] = h[i];
  __syncthreads();
  const int j4 = (blockIdx.x * blockDim.x + threadIdx.x) * 4;
  if (j4 >= NOUT) return;
  float4 acc = *(const float4*)(b2 + j4);
  const float* wp = W2 + j4;
  for (int k = 0; k < HID; ++k) {
    const float  hk = hl[k];
    const float4 w  = *(const float4*)wp;
    __builtin_prefetch(wp + 4 * (size_t)NOUT, 0, 1);   // global_prefetch a few rows ahead
    acc.x = fmaf(hk, w.x, acc.x);
    acc.y = fmaf(hk, w.y, acc.y);
    acc.z = fmaf(hk, w.z, acc.z);
    acc.w = fmaf(hk, w.w, acc.w);
    wp += NOUT;
  }
  const int sid = j4 / SEG;
  const int jl  = j4 - sid * SEG;
  uint2 pk;
  pk.x = (unsigned)f2bf(acc.x) | ((unsigned)f2bf(acc.y) << 16);
  pk.y = (unsigned)f2bf(acc.z) | ((unsigned)f2bf(acc.w) << 16);
  *(uint2*)(spad + (size_t)sid * SPAD_STRIDE + PADF + jl) = pk;
}

// ---------------------------------------------------------------
// 5) Gaussian smoothing as dense bf16 WMMA, f32 accumulate.
//    Each wave carries TWO 256-output blocks (i0, i0+256) so one staged
//    A (window) fragment feeds two independent WMMA chains per k-step:
//    3 b128 loads per wmma instead of 4, and two interleavable XDL chains.
//      A[m][c] = wrev[k0 - m + c]           (staged, aligned, lane-major)
//      B[c][n] = spad[i0 + 16n + k0 + c]    (2 aligned uint4 loads/lane)
//      D[m][n] accumulates O[i0 + m + 16n] exactly once per tap.
//    D layout => each lane holds 8 consecutive outputs: 2 float4 stores.
__global__ void k_conv(const unsigned short* __restrict__ stagedA,
                       const unsigned short* __restrict__ spad,
                       float* __restrict__ out) {
  const int sid  = blockIdx.x / 20;
  const int grp  = blockIdx.x % 20;
  const int wave = threadIdx.x >> 5;
  const int lane = threadIdx.x & 31;
  const int i0   = grp * 1024 + wave * 512;     // <= 19968: every wave fully active
  const unsigned short* Aseg = stagedA + (size_t)sid * (KITER * 32 * 16);
  const unsigned short* sp   = spad    + (size_t)sid * SPAD_STRIDE;
  const int n  = lane & 15;
  const int hi = lane >> 4;
  const int bn = i0 + (n << 4) + (hi << 4);     // B lane base (mult. of 16 -> 32B aligned)
  v8f acc0 = {};
  v8f acc1 = {};
  for (int j = 0; j < KITER; ++j) {
    BF16x16 a, b0, b1;
    const uint4* ap = (const uint4*)(Aseg + ((size_t)(j * 32 + lane) << 4));
    a.u[0] = ap[0];
    a.u[1] = ap[1];
    const uint4* bp0 = (const uint4*)(sp + bn + (j << 5));
    b0.u[0] = bp0[0];
    b0.u[1] = bp0[1];
    const uint4* bp1 = (const uint4*)(sp + bn + 256 + (j << 5));
    b1.u[0] = bp1[0];
    b1.u[1] = bp1[1];
    acc0 = __builtin_amdgcn_wmma_f32_16x16x32_bf16(
        false, a.v, false, b0.v, (short)0, acc0, false, false);
    acc1 = __builtin_amdgcn_wmma_f32_16x16x32_bf16(
        false, a.v, false, b1.v, (short)0, acc1, false, false);
  }
  const int ob0 = i0 + (n << 4) + (hi << 3);    // 8 consecutive outputs per lane
  const int ob1 = ob0 + 256;
  float* oseg = out + (size_t)sid * SEG;
  if (ob0 + 8 <= SEG) {
    F32x8 r; r.v = acc0;
    float4* op = (float4*)(oseg + ob0);
    op[0] = r.f4[0];
    op[1] = r.f4[1];
  }
  if (ob1 + 8 <= SEG) {
    F32x8 r; r.v = acc1;
    float4* op = (float4*)(oseg + ob1);
    op[0] = r.f4[0];
    op[1] = r.f4[1];
  }
}

// ---------------------------------------------------------------
extern "C" void kernel_launch(void* const* d_in, const int* in_sizes, int n_in,
                              void* d_out, int out_size, void* d_ws, size_t ws_size,
                              hipStream_t stream) {
  const float* x     = (const float*)d_in[0];
  const float* W_sig = (const float*)d_in[1];
  const float* b_sig = (const float*)d_in[2];
  const float* W1    = (const float*)d_in[3];
  const float* b1    = (const float*)d_in[4];
  const float* W2    = (const float*)d_in[5];
  const float* b2    = (const float*)d_in[6];

  char* ws = (char*)d_ws;
  float*          wsH     = (float*)(ws + WS_H);
  float*          wsSig   = (float*)(ws + WS_SIG);
  float*          wsSum   = (float*)(ws + WS_WSUM);
  unsigned short* spad    = (unsigned short*)(ws + WS_SPAD);
  unsigned short* stagedA = (unsigned short*)(ws + WS_STAGEA);
  float*          out     = (float*)d_out;

  const int nzero32 = NSEG * SPAD_STRIDE / 2;                 // u16 pairs
  k_zero  <<<(nzero32 + 255) / 256, 256, 0, stream>>>((unsigned int*)spad, nzero32);
  k_head  <<<1, 256, 0, stream>>>(x, W_sig, b_sig, W1, b1, wsH, wsSig);
  k_wsum  <<<NSEG, 256, 0, stream>>>(wsSig, wsSum);
  k_stageA<<<(NSEG * KITER * 32 + 255) / 256, 256, 0, stream>>>(wsSig, wsSum, stagedA);
  k_gemv  <<<(NOUT / 4 + 127) / 128, 128, 0, stream>>>(wsH, W2, b2, spad);
  k_conv  <<<NSEG * 20, 64, 0, stream>>>(stagedA, spad, out);
}
